// MultiHeadAttention_69492570849680
// MI455X (gfx1250) — matrix operand order repaired and checked by reading, not compiled, not run on hardware
//
#include <hip/hip_runtime.h>
#include <hip/hip_bf16.h>
#include <stdint.h>
#include <stddef.h>

// ---------------- types / constants ----------------
typedef __attribute__((ext_vector_type(16))) __bf16 bf16x16;
typedef __attribute__((ext_vector_type(8)))  __bf16 bf16x8;
typedef __attribute__((ext_vector_type(8)))  float  f32x8;

#define N_EMBD   1024
#define N_HEAD   16
#define HEAD_DIM 64
#define SEQ_T    2048
#define BATCH    4
#define BT_ROWS  (BATCH * SEQ_T)   // 8192

#define WMMA_BF16(A, B, C) \
  __builtin_amdgcn_wmma_f32_16x16x32_bf16(false, (A), false, (B), (short)0, (C), false, false)

// A-fragment (16x32 bf16): lane L holds row M=L%16; lanes<16: K 0..15; lanes>=16: K 16..31.
// Caller bakes in row offset and klo = (lane<16 ? 0 : 16); p points at element K = kbase + klo.
static __device__ __forceinline__ bf16x16 load_a_frag(const __bf16* p) {
  return *(const bf16x16*)(p);
}

// B-fragment (32x16 bf16): lane L holds column N=L%16; 16 contiguous K values,
// lanes<16: K 0..15, lanes>=16: K 16..31. Caller bakes in +16 for hi lanes.
static __device__ __forceinline__ bf16x16 load_b_frag(const __bf16* p) {
  return *(const bf16x16*)(p);
}

// CDNA5 async global->LDS copy, 16 bytes per lane, tracked by ASYNCcnt.
// LDS destination address = wave-relative offset = low 32 bits of the generic pointer
// (ISA 10.2: generic LDS addresses carry the LDS offset in addr[31:0]).
static __device__ __forceinline__ void async_g2l_b128(const __bf16* g, __bf16* l) {
  unsigned lds_off = (unsigned)(uintptr_t)l;
  asm volatile("global_load_async_to_lds_b128 %0, %1, off"
               :: "v"(lds_off), "v"(g) : "memory");
}
static __device__ __forceinline__ void wait_asynccnt0() {
  asm volatile("s_wait_asynccnt 0" ::: "memory");
}

// ---------------- convert / transpose helpers ----------------
__global__ void f32_to_bf16_kernel(const float* __restrict__ s, __bf16* __restrict__ d, int n) {
  int i = blockIdx.x * blockDim.x + threadIdx.x;
  if (i < n) d[i] = (__bf16)s[i];
}

// s is [K][N] f32 row-major -> d is [N][K] bf16 row-major
__global__ void transpose_f32_to_bf16_kernel(const float* __restrict__ s, __bf16* __restrict__ d,
                                             int K, int N) {
  int i = blockIdx.x * blockDim.x + threadIdx.x;
  if (i < K * N) {
    int k = i / N, n = i - k * N;
    d[(size_t)n * K + k] = (__bf16)s[i];
  }
}

// ---------------- bf16 WMMA GEMM with async-LDS double buffering ----------------
// C[M][N] = A[M][K] @ B[K][N] + bias, with B given transposed (BT_[N][K]).
// mode 0: QKV epilogue -> scatter bf16 into Q (scaled by 1/8), K, and transposed V buffers.
// mode 1: plain f32 output (final projection).
// Block: 256 threads = 8 waves as 2(M) x 4(N); block tile 128x128; wave tile 64x32.
// Per K-step (32): A/B 128x32 bf16 tiles staged into LDS via global_load_async_to_lds_b128.
__global__ __launch_bounds__(256) void gemm_bf16_wmma(
    const __bf16* __restrict__ A, const __bf16* __restrict__ BT_,
    const float* __restrict__ bias, float* __restrict__ outF,
    __bf16* __restrict__ qbuf, __bf16* __restrict__ kbuf, __bf16* __restrict__ vbufT,
    int M, int N, int K, int mode) {
  __shared__ __bf16 Asmem[2][128][32];   // 8 KB per buffer
  __shared__ __bf16 Bsmem[2][128][32];   // 8 KB per buffer  (rows are N-columns of B)

  const int nb   = N >> 7;
  const int bm   = blockIdx.x / nb;
  const int bn   = blockIdx.x - bm * nb;
  const int tid  = threadIdx.x;
  const int lane = tid & 31;
  const int wave = tid >> 5;
  const int lm   = lane & 15;
  const bool hiL = lane >= 16;
  const int kloA = hiL ? 16 : 0;
  const int kloB = hiL ? 16 : 0;
  const int rw0  = (wave >> 2) * 64;   // wave M offset within block tile
  const int cw0  = (wave & 3) * 32;    // wave N offset within block tile

  // Staging map: thread t covers row = t>>1 (0..127), 16-element half = (t&1)*16.
  const int srow  = tid >> 1;
  const int scol  = (tid & 1) * 16;
  const __bf16* gA = A   + (size_t)(bm * 128 + srow) * K + scol;
  const __bf16* gB = BT_ + (size_t)(bn * 128 + srow) * K + scol;

  f32x8 acc[4][2];
#pragma unroll
  for (int i = 0; i < 4; ++i)
#pragma unroll
    for (int j = 0; j < 2; ++j)
#pragma unroll
      for (int r = 0; r < 8; ++r) acc[i][j][r] = 0.f;

  // prologue: stage K-tile 0 into buffer 0
  async_g2l_b128(gA,     &Asmem[0][srow][scol]);
  async_g2l_b128(gA + 8, &Asmem[0][srow][scol + 8]);
  async_g2l_b128(gB,     &Bsmem[0][srow][scol]);
  async_g2l_b128(gB + 8, &Bsmem[0][srow][scol + 8]);
  wait_asynccnt0();
  __syncthreads();

  for (int k0 = 0; k0 < K; k0 += 32) {
    const int cur = (k0 >> 5) & 1;
    const int nxt = cur ^ 1;
    if (k0 + 32 < K) {  // stage next K-tile while computing on current
      const int kn = k0 + 32;
      async_g2l_b128(gA + kn,     &Asmem[nxt][srow][scol]);
      async_g2l_b128(gA + kn + 8, &Asmem[nxt][srow][scol + 8]);
      async_g2l_b128(gB + kn,     &Bsmem[nxt][srow][scol]);
      async_g2l_b128(gB + kn + 8, &Bsmem[nxt][srow][scol + 8]);
      if (k0 + 64 < K) {  // warm GL2 for the tile after next
        __builtin_prefetch(gA + k0 + 64, 0, 3);
        __builtin_prefetch(gB + k0 + 64, 0, 3);
      }
    }
    bf16x16 af[4], bf[2];
#pragma unroll
    for (int i = 0; i < 4; ++i) af[i] = load_a_frag(&Asmem[cur][rw0 + i * 16 + lm][kloA]);
#pragma unroll
    for (int j = 0; j < 2; ++j) bf[j] = load_b_frag(&Bsmem[cur][cw0 + j * 16 + lm][kloB]);
#pragma unroll
    for (int i = 0; i < 4; ++i)
#pragma unroll
      for (int j = 0; j < 2; ++j) acc[i][j] = WMMA_BF16(af[i], bf[j], acc[i][j]);
    wait_asynccnt0();   // own async stores to LDS complete...
    __syncthreads();    // ...and everyone's, before next iteration reads/overwrites
  }

  // Epilogue. C/D layout: lanes 0-15 col=lm rows r; lanes 16-31 col=lm rows r+8.
  const int rAdd = hiL ? 8 : 0;
  const int row0 = bm * 128 + rw0;
  const int col0 = bn * 128 + cw0;
#pragma unroll
  for (int i = 0; i < 4; ++i)
#pragma unroll
    for (int j = 0; j < 2; ++j) {
      const int col = col0 + j * 16 + lm;
      const float bv = bias[col];
#pragma unroll
      for (int r = 0; r < 8; ++r) {
        const int row = row0 + i * 16 + r + rAdd;
        const float v = acc[i][j][r] + bv;
        if (mode == 1) {
          outF[(size_t)row * N + col] = v;
        } else {
          const int b = row >> 11;        // row / SEQ_T
          const int t = row & (SEQ_T - 1);
          if (col < N_EMBD) {             // Q, pre-scaled by 1/sqrt(Dh)
            const int h = col >> 6, d = col & 63;
            qbuf[(((size_t)(b * N_HEAD + h)) * SEQ_T + t) * HEAD_DIM + d] = (__bf16)(v * 0.125f);
          } else if (col < 2 * N_EMBD) {  // K
            const int c = col - N_EMBD;
            const int h = c >> 6, d = c & 63;
            kbuf[(((size_t)(b * N_HEAD + h)) * SEQ_T + t) * HEAD_DIM + d] = (__bf16)v;
          } else {                        // V stored transposed: [B*H][Dh][T]
            const int c = col - 2 * N_EMBD;
            const int h = c >> 6, d = c & 63;
            vbufT[(((size_t)(b * N_HEAD + h)) * HEAD_DIM + d) * SEQ_T + t] = (__bf16)v;
          }
        }
      }
    }
}

// ---------------- fused causal flash attention ----------------
// Q [B*H][T][64] (pre-scaled), K [B*H][T][64], VT [B*H][64][T]; Y out bf16 [B][T][C].
// 8 waves/block, each wave owns 16 q rows => 128 rows per block; grid = B*H*(T/128).
__global__ __launch_bounds__(256) void attn_flash_wmma(
    const __bf16* __restrict__ Q, const __bf16* __restrict__ Kb,
    const __bf16* __restrict__ VT, __bf16* __restrict__ Y) {
  __shared__ __bf16 plds[8][16][HEAD_DIM];  // per-wave P tile (C-layout -> A-layout staging)

  const int lane = threadIdx.x & 31;
  const int wave = threadIdx.x >> 5;
  const int lm   = lane & 15;
  const bool hiL = lane >= 16;
  const int kloA = hiL ? 16 : 0;
  const int kloB = hiL ? 16 : 0;
  const int rAdd = hiL ? 8 : 0;

  const int bh   = blockIdx.x >> 4;     // T/128 = 16 q-blocks per (b,h)
  const int qblk = blockIdx.x & 15;
  const int qw   = qblk * 128 + wave * 16;
  const int b    = bh >> 4;
  const int h    = bh & 15;

  const __bf16* Qp = Q  + (size_t)bh * SEQ_T * HEAD_DIM;
  const __bf16* Kp = Kb + (size_t)bh * SEQ_T * HEAD_DIM;
  const __bf16* Vp = VT + (size_t)bh * HEAD_DIM * SEQ_T;

  // Q A-fragments for this wave's 16 rows (Dh=64 -> 2 K-steps of 32)
  bf16x16 qa[2];
#pragma unroll
  for (int s = 0; s < 2; ++s)
    qa[s] = load_a_frag(Qp + (size_t)(qw + lm) * HEAD_DIM + s * 32 + kloA);

  float m[8], l[8], alpha[8];
  f32x8 o[4];
#pragma unroll
  for (int r = 0; r < 8; ++r) { m[r] = -3.0e38f; l[r] = 0.f; }
#pragma unroll
  for (int t = 0; t < 4; ++t)
#pragma unroll
    for (int r = 0; r < 8; ++r) o[t][r] = 0.f;

  for (int kb = 0; kb <= qw; kb += 64) {     // causal: last block <= qw (qw = 16-multiple)
    // S = Q @ K^T : 16 x 64 scores as 4 C-tiles
    f32x8 st[4];
#pragma unroll
    for (int t = 0; t < 4; ++t)
#pragma unroll
      for (int r = 0; r < 8; ++r) st[t][r] = 0.f;
#pragma unroll
    for (int nt = 0; nt < 4; ++nt) {
#pragma unroll
      for (int s = 0; s < 2; ++s) {
        bf16x16 kf = load_b_frag(Kp + (size_t)(kb + nt * 16 + lm) * HEAD_DIM + s * 32 + kloB);
        st[nt] = WMMA_BF16(qa[s], kf, st[nt]);
      }
    }
    // causal mask: only the final key block can cross the diagonal
    if (kb + 63 > qw) {
#pragma unroll
      for (int t = 0; t < 4; ++t) {
        const int col = kb + t * 16 + lm;
#pragma unroll
        for (int r = 0; r < 8; ++r)
          if (col > qw + r + rAdd) st[t][r] = -3.0e38f;
      }
    }
    // online softmax (row stats via intra-half-wave shuffles)
#pragma unroll
    for (int r = 0; r < 8; ++r) {
      float mx = fmaxf(fmaxf(st[0][r], st[1][r]), fmaxf(st[2][r], st[3][r]));
#pragma unroll
      for (int off = 8; off >= 1; off >>= 1) mx = fmaxf(mx, __shfl_xor(mx, off, 32));
      const float mn = fmaxf(m[r], mx);
      alpha[r] = __expf(m[r] - mn);
      m[r] = mn;
      float rs = 0.f;
#pragma unroll
      for (int t = 0; t < 4; ++t) {
        const float p = __expf(st[t][r] - mn);
        st[t][r] = p;
        rs += p;
      }
#pragma unroll
      for (int off = 8; off >= 1; off >>= 1) rs += __shfl_xor(rs, off, 32);
      l[r] = l[r] * alpha[r] + rs;
    }
    // rescale O, spill P (bf16) to LDS in C-layout
#pragma unroll
    for (int t = 0; t < 4; ++t)
#pragma unroll
      for (int r = 0; r < 8; ++r) {
        o[t][r] *= alpha[r];
        plds[wave][r + rAdd][t * 16 + lm] = (__bf16)st[t][r];
      }
    // wave-private LDS tile, but the write->read dependency is cross-lane:
    // DS ops are in-order within a wave; stop compiler reordering + drain DScnt.
    asm volatile("s_wait_dscnt 0" ::: "memory");
    // O += P @ V : P read back in A-layout, V columns are contiguous rows of VT
#pragma unroll
    for (int s = 0; s < 2; ++s) {
      bf16x16 pf = load_a_frag(&plds[wave][lm][s * 32 + kloA]);
#pragma unroll
      for (int dt = 0; dt < 4; ++dt) {
        bf16x16 vf = load_b_frag(Vp + (size_t)(dt * 16 + lm) * SEQ_T + kb + s * 32 + kloB);
        o[dt] = WMMA_BF16(pf, vf, o[dt]);
      }
    }
    asm volatile("s_wait_dscnt 0" ::: "memory");  // before next iteration overwrites plds
  }

  // normalize and write y[b][t][h*64+d] as bf16 for the projection GEMM
#pragma unroll
  for (int dt = 0; dt < 4; ++dt)
#pragma unroll
    for (int r = 0; r < 8; ++r) {
      const int trow = qw + r + rAdd;
      const float v = o[dt][r] / l[r];
      Y[((size_t)(b * SEQ_T + trow)) * N_EMBD + h * HEAD_DIM + dt * 16 + lm] = (__bf16)v;
    }
}

// ---------------- host launcher ----------------
extern "C" void kernel_launch(void* const* d_in, const int* in_sizes, int n_in,
                              void* d_out, int out_size, void* d_ws, size_t ws_size,
                              hipStream_t stream) {
  (void)in_sizes; (void)n_in; (void)out_size; (void)ws_size;
  const float* x      = (const float*)d_in[0];  // [B,T,C]
  const float* w_qkv  = (const float*)d_in[1];  // [C,3C]
  const float* b_qkv  = (const float*)d_in[2];  // [3C]
  const float* w_proj = (const float*)d_in[3];  // [C,C]
  const float* b_proj = (const float*)d_in[4];  // [C]
  float* out = (float*)d_out;                   // [B,T,C] f32

  char* ws = (char*)d_ws;
  size_t off = 0;
  auto take = [&](size_t bytes) -> char* {
    char* p = ws + off;
    off += (bytes + 255) & ~(size_t)255;
    return p;
  };
  __bf16* xh     = (__bf16*)take((size_t)BT_ROWS * N_EMBD * 2);       // 16 MiB
  __bf16* wqkvT  = (__bf16*)take((size_t)3 * N_EMBD * N_EMBD * 2);    //  6 MiB
  __bf16* wprojT = (__bf16*)take((size_t)N_EMBD * N_EMBD * 2);        //  2 MiB
  __bf16* qbuf   = (__bf16*)take((size_t)BT_ROWS * N_EMBD * 2);       // 16 MiB
  __bf16* kbuf   = (__bf16*)take((size_t)BT_ROWS * N_EMBD * 2);       // 16 MiB
  __bf16* vbufT  = (__bf16*)take((size_t)BT_ROWS * N_EMBD * 2);       // 16 MiB
  __bf16* ybuf   = (__bf16*)take((size_t)BT_ROWS * N_EMBD * 2);       // 16 MiB

  // 1) precision conversion / weight transposition
  {
    const int n = BT_ROWS * N_EMBD;
    f32_to_bf16_kernel<<<n / 256, 256, 0, stream>>>(x, xh, n);
  }
  transpose_f32_to_bf16_kernel<<<(N_EMBD * 3 * N_EMBD) / 256, 256, 0, stream>>>(
      w_qkv, wqkvT, N_EMBD, 3 * N_EMBD);
  transpose_f32_to_bf16_kernel<<<(N_EMBD * N_EMBD) / 256, 256, 0, stream>>>(
      w_proj, wprojT, N_EMBD, N_EMBD);

  // 2) QKV projection GEMM (scatter into Q scaled, K, V^T)
  gemm_bf16_wmma<<<(BT_ROWS / 128) * (3 * N_EMBD / 128), 256, 0, stream>>>(
      xh, wqkvT, b_qkv, nullptr, qbuf, kbuf, vbufT,
      BT_ROWS, 3 * N_EMBD, N_EMBD, /*mode=*/0);

  // 3) fused causal flash attention
  attn_flash_wmma<<<BATCH * N_HEAD * (SEQ_T / 128), 256, 0, stream>>>(qbuf, kbuf, vbufT, ybuf);

  // 4) output projection GEMM (f32 out)
  gemm_bf16_wmma<<<(BT_ROWS / 128) * (N_EMBD / 128), 256, 0, stream>>>(
      ybuf, wprojT, b_proj, out, nullptr, nullptr, nullptr,
      BT_ROWS, N_EMBD, N_EMBD, /*mode=*/1);
}
